// DQN_30167850287770
// MI455X (gfx1250) — matrix operand order repaired and checked by reading, not compiled, not run on hardware
//
#include <hip/hip_runtime.h>
#include <hip/hip_bf16.h>

// ---------------------------------------------------------------------------
// GAT + MLP pipeline for MI455X (gfx1250, wave32, WMMA).
// Heavy GEMMs use v_wmma_f32_16x16x32_f16 (f16 in, f32 accumulate) with
// LDS-pre-swizzled B fragments (per-lane-contiguous -> 2x ds_load_b128).
// Edge softmax/aggregation are bandwidth/atomic bound -> VALU + atomics.
// ---------------------------------------------------------------------------

#define NN 200000
#define EE 3200000
#define BB 1024
#define AA 10
#define NEG_SLOPE 0.2f

typedef __attribute__((ext_vector_type(16))) _Float16 v16h;
typedef __attribute__((ext_vector_type(8)))  float    v8f;

__device__ __forceinline__ v8f wmma_f16(v16h a, v16h b, v8f c) {
    return __builtin_amdgcn_wmma_f32_16x16x32_f16(
        /*neg_a=*/false, a, /*neg_b=*/false, b,
        /*c_mod=*/(short)0, c, /*reuse_a=*/false, /*reuse_b=*/false);
}

// order-preserving float <-> signed-int key (for atomicMax segment-max)
__device__ __forceinline__ int   fkey(float f) { int i = __float_as_int(f); return i >= 0 ? i : (i ^ 0x7FFFFFFF); }
__device__ __forceinline__ float funkey(int i) { return __int_as_float(i >= 0 ? i : (i ^ 0x7FFFFFFF)); }

__device__ __forceinline__ float lrelu(float f) { return f >= 0.f ? f : NEG_SLOPE * f; }

// --------------------------- tiny scalar kernels ---------------------------

// sedge = dot(W_edge[64], att_edge[64])   (ED == 1)
__global__ void k_sdot(const float* __restrict__ We, const float* __restrict__ ae,
                       float* __restrict__ sed) {
    __shared__ float red[64];
    int t = threadIdx.x;
    red[t] = We[t] * ae[t];
    __syncthreads();
    for (int s = 32; s > 0; s >>= 1) { if (t < s) red[t] += red[t + s]; __syncthreads(); }
    if (t == 0) sed[0] = red[0];
}

__global__ void k_init(float* __restrict__ deg, float* __restrict__ asum,
                       float* __restrict__ psum, float* __restrict__ cnt) {
    int i = blockIdx.x * blockDim.x + threadIdx.x;
    if (i < NN)       { deg[i] = 0.f; asum[i] = 0.f; }
    if (i < BB * 128) { psum[i] = 0.f; }
    if (i < BB)       { cnt[i] = 0.f; }
}

// in-degree and edge_attr sum per destination (for self-loop fill 'mean')
__global__ void k_deg(const int* __restrict__ ei, const float* __restrict__ eattr,
                      float* __restrict__ deg, float* __restrict__ asum) {
    int e = blockIdx.x * blockDim.x + threadIdx.x;
    if (e >= EE) return;
    int d = ei[EE + e];
    atomicAdd(&deg[d], 1.f);
    atomicAdd(&asum[d], eattr[e]);
}

// ------------------------------ GEMM1 (WMMA) -------------------------------
// h[N,64] = x[N,9] @ W_gat[9,64]; K padded 9 -> 32 with zeros.
// B fragments pre-swizzled in LDS: group (tc) x lane -> 16 contiguous halves.
__global__ void k_gemm1(const float* __restrict__ x, const float* __restrict__ Wg,
                        float* __restrict__ h) {
    __shared__ __align__(32) _Float16 swz[4 * 32 * 16];   // 4 KB
    int t = threadIdx.x;
    for (int i = t; i < 4 * 32 * 16; i += 256) swz[i] = (_Float16)0.0f;
    __syncthreads();
    // fill: element W_gat[r][c] -> tile tc=c>>4, lane=(c&15)+16*(r>>3), j=r&7
    for (int i = t; i < 9 * 64; i += 256) {
        int r = i >> 6, c = i & 63;
        swz[((c >> 4) * 32 + (c & 15) + (r >> 3) * 16) * 16 + (r & 7)] = (_Float16)Wg[i];
    }
    __syncthreads();
    int wave = t >> 5, lane = t & 31;
    int tileBase = (blockIdx.x * 8 + wave) * 16;
    if (tileBase >= NN) return;
    bool lo = lane < 16;
    int mrow = lane & 15;

    // A fragment (ISA 16-bit A layout: lo lanes K0-7/16-23, hi lanes K8-15/24-31)
    v16h a = {};
    int row = tileBase + mrow;
    if (lo) {
        const float* xr = x + (size_t)row * 9;
#pragma unroll
        for (int j = 0; j < 8; ++j) a[j] = (_Float16)xr[j];     // K = 0..7
    } else {
        a[0] = (_Float16)x[(size_t)row * 9 + 8];                // K = 8
    }
#pragma unroll
    for (int tc = 0; tc < 4; ++tc) {
        v16h b = *(const v16h*)&swz[(tc * 32 + lane) * 16];     // 2x ds_load_b128
        v8f c = {};
        c = wmma_f16(a, b, c);
        int nIdx = lane & 15;
        int mBase = lo ? 0 : 8;
        float* hp = h + (size_t)(tileBase + mBase) * 64 + tc * 16 + nIdx;
#pragma unroll
        for (int r = 0; r < 8; ++r) hp[(size_t)r * 64] = c[r];
    }
}

// -------------------------- per-node attention prep ------------------------
__global__ void k_node_att(const float* __restrict__ h,
                           const float* __restrict__ att_src, const float* __restrict__ att_dst,
                           const float* __restrict__ sed,
                           const float* __restrict__ deg, const float* __restrict__ asum,
                           const int* __restrict__ pool_batch,
                           float* __restrict__ a_src, float* __restrict__ a_dst,
                           float* __restrict__ selfbuf, int* __restrict__ m,
                           float* __restrict__ cnt) {
    __shared__ float sAs[64], sAd[64];
    int t = threadIdx.x;
    if (t < 64) { sAs[t] = att_src[t]; sAd[t] = att_dst[t]; }
    __syncthreads();
    int i = blockIdx.x * blockDim.x + t;
    if (i >= NN) return;
    const float4* hr = (const float4*)(h + (size_t)i * 64);
    float as = 0.f, ad = 0.f;
#pragma unroll
    for (int k = 0; k < 16; ++k) {
        float4 v = hr[k];
        int k4 = k * 4;
        as += v.x * sAs[k4] + v.y * sAs[k4 + 1] + v.z * sAs[k4 + 2] + v.w * sAs[k4 + 3];
        ad += v.x * sAd[k4] + v.y * sAd[k4 + 1] + v.z * sAd[k4 + 2] + v.w * sAd[k4 + 3];
    }
    a_src[i] = as; a_dst[i] = ad;
    float loop_attr = asum[i] / fmaxf(deg[i], 1.f);
    float al = lrelu(as + ad + sed[0] * loop_attr);   // self-loop alpha
    selfbuf[i] = al;
    m[i] = fkey(al);                                  // init segment-max with self edge
    atomicAdd(&cnt[pool_batch[i]], 1.f);
}

// per-edge alpha + segment max (via order-preserving int key atomicMax)
__global__ void k_edge_alpha(const int* __restrict__ ei, const float* __restrict__ eattr,
                             const float* __restrict__ a_src, const float* __restrict__ a_dst,
                             const float* __restrict__ sed,
                             float* __restrict__ abuf, int* __restrict__ m) {
    int e = blockIdx.x * blockDim.x + threadIdx.x;
    if (e >= EE) return;
    int s = ei[e], d = ei[EE + e];
    float al = lrelu(a_src[s] + a_dst[d] + sed[0] * eattr[e]);
    abuf[e] = al;
    atomicMax(&m[d], fkey(al));
}

// decode max, self-loop exp, init denom
__global__ void k_node_soft(int* __restrict__ m, float* __restrict__ selfbuf,
                            float* __restrict__ denom) {
    int i = blockIdx.x * blockDim.x + threadIdx.x;
    if (i >= NN) return;
    float mf = funkey(m[i]);
    ((float*)m)[i] = mf;                      // reuse buffer as float max
    float ea = __expf(selfbuf[i] - mf);
    selfbuf[i] = ea;
    denom[i] = ea;
}

__global__ void k_edge_exp(const int* __restrict__ ei, float* __restrict__ abuf,
                           const float* __restrict__ mf, float* __restrict__ denom) {
    int e = blockIdx.x * blockDim.x + threadIdx.x;
    if (e >= EE) return;
    int d = ei[EE + e];
    float ea = __expf(abuf[e] - mf[d]);
    abuf[e] = ea;
    atomicAdd(&denom[d], ea);
}

// accum[i,:] = (self coef) * h[i,:]   (plain store; edge atomics come after)
__global__ void k_self_agg(const float* __restrict__ h, const float* __restrict__ selfbuf,
                           const float* __restrict__ denom, float* __restrict__ accum) {
    size_t idx = (size_t)blockIdx.x * blockDim.x + threadIdx.x;
    if (idx >= (size_t)NN * 64) return;
    int i = (int)(idx >> 6);
    accum[idx] = (selfbuf[i] / denom[i]) * h[idx];
}

// one wave per edge, 2 columns per lane: accum[dst,:] += coef * h[src,:]
__global__ void k_edge_agg(const int* __restrict__ ei, const float* __restrict__ abuf,
                           const float* __restrict__ denom, const float* __restrict__ h,
                           float* __restrict__ accum) {
    size_t tid = (size_t)blockIdx.x * blockDim.x + threadIdx.x;
    int e = (int)(tid >> 5);
    if (e >= EE) return;
    int lane = (int)(tid & 31);
    int s = ei[e], d = ei[EE + e];
    float coef = abuf[e] / denom[d];
    const float* hs = h + (size_t)s * 64;
    float* ad = accum + (size_t)d * 64;
    atomicAdd(&ad[lane],      coef * hs[lane]);
    atomicAdd(&ad[lane + 32], coef * hs[lane + 32]);
}

// ------------------------------ GEMM2 (WMMA) -------------------------------
// g = relu((accum + b_gat) @ W1 + b1); fused global-mean-pool accumulation.
// W1 pre-swizzled: group (tc*2+s) x lane -> 16 contiguous halves (bijection).
__global__ void k_gemm2(const float* __restrict__ accum, const float* __restrict__ bgat,
                        const float* __restrict__ W1, const float* __restrict__ b1,
                        const int* __restrict__ pool_batch, float* __restrict__ psum) {
    __shared__ __align__(32) _Float16 swz[16 * 32 * 16];  // 16 KB
    __shared__ float sbg[64];
    __shared__ float sb1[128];
    int t = threadIdx.x;
    for (int i = t; i < 64 * 128; i += 256) {
        int k = i >> 7, col = i & 127;
        int tc = col >> 4, nIdx = col & 15;
        int s = k >> 5, kr = k & 31;
        int lane = nIdx + ((kr & 8) ? 16 : 0);
        swz[((tc * 2 + s) * 32 + lane) * 16 + ((kr & 7) | ((kr & 16) >> 1))] = (_Float16)W1[i];
    }
    if (t < 64)  sbg[t] = bgat[t];
    if (t < 128) sb1[t] = b1[t];
    __syncthreads();
    int wave = t >> 5, lane = t & 31;
    int tileBase = (blockIdx.x * 8 + wave) * 16;
    if (tileBase >= NN) return;
    bool lo = lane < 16;
    int mrow = lane & 15;
    const float4* ar4 = (const float4*)(accum + (size_t)(tileBase + mrow) * 64);

    // A fragments for the two K=32 steps (float4 global loads)
    v16h a[2];
#pragma unroll
    for (int s = 0; s < 2; ++s) {
        int k0 = s * 32 + (lo ? 0 : 8);
        float4 u0 = ar4[(k0 >> 2)], u1 = ar4[(k0 >> 2) + 1];
        float4 v0 = ar4[(k0 >> 2) + 4], v1 = ar4[(k0 >> 2) + 5];
        float c0[8] = {u0.x, u0.y, u0.z, u0.w, u1.x, u1.y, u1.z, u1.w};
        float c1[8] = {v0.x, v0.y, v0.z, v0.w, v1.x, v1.y, v1.z, v1.w};
#pragma unroll
        for (int j = 0; j < 8; ++j) a[s][j]     = (_Float16)(c0[j] + sbg[k0 + j]);
#pragma unroll
        for (int j = 0; j < 8; ++j) a[s][8 + j] = (_Float16)(c1[j] + sbg[k0 + 16 + j]);
    }
    int nIdx = lane & 15;
    int mBase = lo ? 0 : 8;
    int batch[8];
#pragma unroll
    for (int r = 0; r < 8; ++r) batch[r] = pool_batch[tileBase + mBase + r];

#pragma unroll
    for (int tc = 0; tc < 8; ++tc) {
        int col = tc * 16 + nIdx;
        v8f c = {};
#pragma unroll
        for (int s = 0; s < 2; ++s) {
            v16h b = *(const v16h*)&swz[((tc * 2 + s) * 32 + lane) * 16]; // 2x ds_load_b128
            c = wmma_f16(a[s], b, c);
        }
        float bb = sb1[col];
#pragma unroll
        for (int r = 0; r < 8; ++r) {
            float v = c[r] + bb;
            v = v > 0.f ? v : 0.f;
            atomicAdd(&psum[batch[r] * 128 + col], v);
        }
    }
}

// ------------------------------- MLP head ----------------------------------
__global__ void k_agent(const float* __restrict__ as, const float* __restrict__ W2,
                        const float* __restrict__ b2, float* __restrict__ a2) {
    int idx = blockIdx.x * blockDim.x + threadIdx.x;
    if (idx >= BB * 64) return;
    int b = idx >> 6, j = idx & 63;
    float acc = b2[j];
    for (int k = 0; k < 34; ++k) acc += as[b * 34 + k] * W2[k * 64 + j];
    a2[idx] = acc > 0.f ? acc : 0.f;
}

__global__ void k_pooldiv(const float* __restrict__ psum, const float* __restrict__ cnt,
                          float* __restrict__ pooled) {
    int idx = blockIdx.x * blockDim.x + threadIdx.x;
    if (idx >= BB * 128) return;
    int b = idx >> 7;
    pooled[idx] = psum[idx] / fmaxf(cnt[b], 1.f);
}

__global__ void k_fc3(const float* __restrict__ pooled, const float* __restrict__ a2,
                      const float* __restrict__ W3, const float* __restrict__ b3,
                      float* __restrict__ z3) {
    int idx = blockIdx.x * blockDim.x + threadIdx.x;
    if (idx >= BB * 128) return;
    int b = idx >> 7, j = idx & 127;
    float acc = b3[j];
    for (int k = 0; k < 128; ++k) acc += pooled[b * 128 + k] * W3[k * 128 + j];
    for (int k = 0; k < 64;  ++k) acc += a2[b * 64 + k]     * W3[(128 + k) * 128 + j];
    z3[idx] = acc > 0.f ? acc : 0.f;
}

__global__ void k_fc4(const float* __restrict__ z3, const float* __restrict__ W4,
                      const float* __restrict__ b4, float* __restrict__ out) {
    int idx = blockIdx.x * blockDim.x + threadIdx.x;
    if (idx >= BB * AA) return;
    int b = idx / AA, j = idx % AA;
    float acc = b4[j];
    for (int k = 0; k < 128; ++k) acc += z3[b * 128 + k] * W4[k * AA + j];
    out[idx] = acc;
}

// ------------------------------- launcher ----------------------------------
extern "C" void kernel_launch(void* const* d_in, const int* in_sizes, int n_in,
                              void* d_out, int out_size, void* d_ws, size_t ws_size,
                              hipStream_t stream) {
    const float* x      = (const float*)d_in[0];
    const int*   ei     = (const int*)  d_in[1];
    const float* eattr  = (const float*)d_in[2];
    const float* agent  = (const float*)d_in[3];
    const int*   pbatch = (const int*)  d_in[4];
    const float* Wg     = (const float*)d_in[5];
    const float* attS   = (const float*)d_in[6];
    const float* attD   = (const float*)d_in[7];
    const float* We     = (const float*)d_in[8];
    const float* attE   = (const float*)d_in[9];
    const float* bgat   = (const float*)d_in[10];
    const float* W1     = (const float*)d_in[11];
    const float* b1     = (const float*)d_in[12];
    const float* W2     = (const float*)d_in[13];
    const float* b2     = (const float*)d_in[14];
    const float* W3     = (const float*)d_in[15];
    const float* b3     = (const float*)d_in[16];
    const float* W4     = (const float*)d_in[17];
    const float* b4     = (const float*)d_in[18];
    float* out = (float*)d_out;

    // workspace carve-up (~121 MB)
    char* ws = (char*)d_ws;
    size_t o = 0;
    auto take = [&](size_t bytes) { char* p = ws + o; o += (bytes + 255) & ~(size_t)255; return p; };
    float* h       = (float*)take((size_t)NN * 64 * 4);
    float* accum   = (float*)take((size_t)NN * 64 * 4);
    float* abuf    = (float*)take((size_t)EE * 4);
    float* a_src   = (float*)take((size_t)NN * 4);
    float* a_dst   = (float*)take((size_t)NN * 4);
    float* deg     = (float*)take((size_t)NN * 4);
    float* asum    = (float*)take((size_t)NN * 4);
    int*   m       = (int*)  take((size_t)NN * 4);
    float* selfbuf = (float*)take((size_t)NN * 4);
    float* denom   = (float*)take((size_t)NN * 4);
    float* psum    = (float*)take((size_t)BB * 128 * 4);
    float* cnt     = (float*)take((size_t)BB * 4);
    float* a2      = (float*)take((size_t)BB * 64 * 4);
    float* pooled  = (float*)take((size_t)BB * 128 * 4);
    float* z3      = (float*)take((size_t)BB * 128 * 4);
    float* sed     = (float*)take(256);
    (void)ws_size; (void)in_sizes; (void)n_in; (void)out_size;

    const int T = 256;
    int gN   = (NN + T - 1) / T;
    int gE   = (EE + T - 1) / T;
    int gN64 = (int)(((size_t)NN * 64 + T - 1) / T);
    int gE32 = (int)(((size_t)EE * 32 + T - 1) / T);
    int gTile = ((NN + 15) / 16 + 7) / 8;   // 16-row tiles, 8 waves/block

    k_sdot<<<1, 64, 0, stream>>>(We, attE, sed);
    k_init<<<gN, T, 0, stream>>>(deg, asum, psum, cnt);
    k_deg<<<gE, T, 0, stream>>>(ei, eattr, deg, asum);
    k_gemm1<<<gTile, T, 0, stream>>>(x, Wg, h);
    k_node_att<<<gN, T, 0, stream>>>(h, attS, attD, sed, deg, asum, pbatch,
                                     a_src, a_dst, selfbuf, m, cnt);
    k_edge_alpha<<<gE, T, 0, stream>>>(ei, eattr, a_src, a_dst, sed, abuf, m);
    k_node_soft<<<gN, T, 0, stream>>>(m, selfbuf, denom);
    k_edge_exp<<<gE, T, 0, stream>>>(ei, abuf, (const float*)m, denom);
    k_self_agg<<<gN64, T, 0, stream>>>(h, selfbuf, denom, accum);
    k_edge_agg<<<gE32, T, 0, stream>>>(ei, abuf, denom, h, accum);
    k_gemm2<<<gTile, T, 0, stream>>>(accum, bgat, W1, b1, pbatch, psum);
    k_agent<<<(BB * 64 + T - 1) / T, T, 0, stream>>>(agent, W2, b2, a2);
    k_pooldiv<<<(BB * 128 + T - 1) / T, T, 0, stream>>>(psum, cnt, pooled);
    k_fc3<<<(BB * 128 + T - 1) / T, T, 0, stream>>>(pooled, a2, W3, b3, z3);
    k_fc4<<<(BB * AA + T - 1) / T, T, 0, stream>>>(z3, W4, b4, out);
}
